// Model_61340722921866
// MI455X (gfx1250) — matrix operand order repaired and checked by reading, not compiled, not run on hardware
//
#include <hip/hip_runtime.h>

typedef __attribute__((ext_vector_type(16))) _Float16 v16h;
typedef __attribute__((ext_vector_type(8)))  _Float16 v8h;
typedef __attribute__((ext_vector_type(8)))  float    v8f;
typedef __attribute__((ext_vector_type(4)))  float    v4f;

#define T_STEPS 512
#define BATCH   1024
#define H       128
#define BM      16          // batch rows per workgroup (one WMMA M-tile)
#define NTHREADS 256        // 8 waves; one (M, gate-column) unit per wave

__device__ __forceinline__ v8f wmma16(v16h a, v16h b, v8f c) {
    // v_wmma_f32_16x16x32_f16: D = A(16x32 f16) * B(32x16 f16) + C(16x16 f32)
    return __builtin_amdgcn_wmma_f32_16x16x32_f16(
        /*neg_a=*/false, a, /*neg_b=*/false, b,
        /*c_mod=*/(short)0, c, /*reuse_a=*/false, /*reuse_b=*/false);
}

// f32 -> v16h (weight B-fragment: K = p[0..7] and p[16..23], table order)
__device__ __forceinline__ v16h cvt_row16(const float* __restrict__ p) {
    v16h r;
#pragma unroll
    for (int i = 0; i < 16; ++i) r[i] = (_Float16)p[i < 8 ? i : i + 8];
    return r;
}

// A-fragment: 8 f16 at p (K = kbase..kbase+7), 8 f16 at p+16 (K = kbase+16..+23)
__device__ __forceinline__ v16h load_a_frag(const _Float16* p) {
    v8h lo = *(const v8h*)(p);
    v8h hi = *(const v8h*)(p + 16);
    return __builtin_shufflevector(lo, hi, 0,1,2,3,4,5,6,7,8,9,10,11,12,13,14,15);
}

__device__ __forceinline__ float fast_sigmoid(float x) {
    return 1.0f / (1.0f + __expf(-x));
}
__device__ __forceinline__ float fast_tanh(float x) {
    float e2 = __expf(2.0f * x);
    return 1.0f - 2.0f / (e2 + 1.0f);
}

__global__ __launch_bounds__(NTHREADS, 1)
void gru_scan_kernel(const float* __restrict__ input,   // [T,B,H]
                     const float* __restrict__ h0,      // [B,H]
                     const float* __restrict__ w_ih,    // [3H,H]
                     const float* __restrict__ w_hh,    // [3H,H]
                     const float* __restrict__ b_ih,    // [3H]
                     const float* __restrict__ b_hh,    // [3H]
                     const float* __restrict__ w_out,   // [1,H]
                     const float* __restrict__ b_out,   // [1]
                     float* __restrict__ out)           // [B,1]
{
    __shared__ __align__(16) _Float16 xbuf[BM * H];   // x[t] tile, f16
    __shared__ __align__(16) _Float16 hbuf[BM * H];   // h tile, f16
    __shared__ __align__(16) float    hf32[BM * H];   // final h, f32
    __shared__ __align__(16) float    woutl[H];

    const int tid  = threadIdx.x;
    const int lane = tid & 31;
    const int wave = tid >> 5;           // 0..7 == gate-column tile
    const int b0   = blockIdx.x * BM;

    const int coltile = wave;            // output columns coltile*16 .. +15
    const int ncol    = lane & 15;
    const int gcol    = coltile * 16 + ncol;        // hidden index 0..127

    // ---- loop-invariant weight B-fragments in registers (f16) ----
    const int kb_b = (lane & 16) ? 8 : 0;           // B layout: table order K / lane
    v16h wr_x[4], wr_h[4], wz_x[4], wz_h[4], wn_x[4], wn_h[4];
    {
        const int nr = 0 * H + gcol;   // gate r row of W
        const int nz = 1 * H + gcol;   // gate z
        const int nn = 2 * H + gcol;   // gate n
#pragma unroll
        for (int kt = 0; kt < 4; ++kt) {
            const int ko = kt * 32 + kb_b;
            wr_x[kt] = cvt_row16(w_ih + nr * H + ko);
            wr_h[kt] = cvt_row16(w_hh + nr * H + ko);
            wz_x[kt] = cvt_row16(w_ih + nz * H + ko);
            wz_h[kt] = cvt_row16(w_hh + nz * H + ko);
            wn_x[kt] = cvt_row16(w_ih + nn * H + ko);
            wn_h[kt] = cvt_row16(w_hh + nn * H + ko);
        }
    }
    // per-lane biases (C-layout: all 8 acc VGPRs share column gcol)
    const float bias_r  = b_ih[gcol]       + b_hh[gcol];
    const float bias_z  = b_ih[H + gcol]   + b_hh[H + gcol];
    const float bias_xn = b_ih[2*H + gcol];
    const float bias_hn = b_hh[2*H + gcol];

    // ---- init LDS: h0 (f16), w_out ----
    for (int i = tid; i < BM * H; i += NTHREADS)
        hbuf[i] = (_Float16)h0[(size_t)b0 * H + i];
    if (tid < H) woutl[tid] = w_out[tid];

    // ---- x prefetch mapping: 8 floats per thread ----
    const int xrow = tid >> 4;            // 0..15
    const int xcol = (tid & 15) * 8;      // 0..120
    v4f pa, pb;
    {   // prefetch x[0]
        const float* p = input + ((size_t)(b0 + xrow)) * H + xcol;
        pa = *(const v4f*)(p);
        pb = *(const v4f*)(p + 4);
    }
    {   // stage x[0] into xbuf, then prefetch x[1]
        v8h xh;
#pragma unroll
        for (int i = 0; i < 4; ++i) { xh[i] = (_Float16)pa[i]; xh[i+4] = (_Float16)pb[i]; }
        *(v8h*)(&xbuf[xrow * H + xcol]) = xh;
        const float* p = input + ((size_t)BATCH * H) + ((size_t)(b0 + xrow)) * H + xcol;
        pa = *(const v4f*)(p);
        pb = *(const v4f*)(p + 4);
    }

    // A-fragment / C-fragment per-lane coordinates
    const int arow  = lane & 15;                 // A row (batch row)
    const int kb_a  = (lane & 16) ? 8 : 0;       // A layout K sub-offset
    const int crow0 = (lane & 16) ? 8 : 0;       // C row base (+v)

    for (int t = 0; t < T_STEPS; ++t) {
        __syncthreads();   // xbuf[t] + current hbuf visible to all waves

        // ---- read phase: A-fragments + h_old (C-layout) ----
        v16h xf[4], hf[4];
#pragma unroll
        for (int kt = 0; kt < 4; ++kt) {
            xf[kt] = load_a_frag(&xbuf[arow * H + kt * 32 + kb_a]);
            hf[kt] = load_a_frag(&hbuf[arow * H + kt * 32 + kb_a]);
        }
        float hold[8];
#pragma unroll
        for (int v = 0; v < 8; ++v)
            hold[v] = (float)hbuf[(crow0 + v) * H + gcol];

        __syncthreads();   // all reads done; safe to overwrite xbuf / hbuf

        // ---- stage x[t+1] (already in regs), prefetch x[t+2] ----
        if (t + 1 < T_STEPS) {
            v8h xh;
#pragma unroll
            for (int i = 0; i < 4; ++i) { xh[i] = (_Float16)pa[i]; xh[i+4] = (_Float16)pb[i]; }
            *(v8h*)(&xbuf[xrow * H + xcol]) = xh;
            if (t + 2 < T_STEPS) {
                const float* p = input + (size_t)(t + 2) * BATCH * H
                               + ((size_t)(b0 + xrow)) * H + xcol;
                pa = *(const v4f*)(p);
                pb = *(const v4f*)(p + 4);
            }
        }

        // ---- GEMMs: 24 x v_wmma_f32_16x16x32_f16 ----
        v8f ar = {}, az = {}, axn = {}, ahn = {};
#pragma unroll
        for (int kt = 0; kt < 4; ++kt) {
            ar  = wmma16(xf[kt], wr_x[kt], ar);    // x·W_ih(r)
            ar  = wmma16(hf[kt], wr_h[kt], ar);    // + h·W_hh(r)
            az  = wmma16(xf[kt], wz_x[kt], az);
            az  = wmma16(hf[kt], wz_h[kt], az);
            axn = wmma16(xf[kt], wn_x[kt], axn);   // x·W_ih(n)  (kept separate)
            ahn = wmma16(hf[kt], wn_h[kt], ahn);   // h·W_hh(n)  (kept separate)
        }

        // ---- gates + h update (per-lane; each lane owns col gcol, rows crow0+v) ----
#pragma unroll
        for (int v = 0; v < 8; ++v) {
            const float r = fast_sigmoid(ar[v] + bias_r);
            const float z = fast_sigmoid(az[v] + bias_z);
            const float n = fast_tanh(axn[v] + bias_xn + r * (ahn[v] + bias_hn));
            const float hn2 = (1.0f - z) * n + z * hold[v];
            hbuf[(crow0 + v) * H + gcol] = (_Float16)hn2;
            if (t == T_STEPS - 1) hf32[(crow0 + v) * H + gcol] = hn2;
        }
    }

    __syncthreads();
    // ---- output projection: out[b] = h·w_out + b_out ----
    if (tid < BM) {
        float acc = b_out[0];
#pragma unroll 8
        for (int j = 0; j < H; ++j) acc += hf32[tid * H + j] * woutl[j];
        out[b0 + tid] = acc;
    }
}

extern "C" void kernel_launch(void* const* d_in, const int* in_sizes, int n_in,
                              void* d_out, int out_size, void* d_ws, size_t ws_size,
                              hipStream_t stream) {
    const float* input = (const float*)d_in[0];
    const float* h0    = (const float*)d_in[1];
    const float* w_ih  = (const float*)d_in[2];
    const float* w_hh  = (const float*)d_in[3];
    const float* b_ih  = (const float*)d_in[4];
    const float* b_hh  = (const float*)d_in[5];
    const float* w_out = (const float*)d_in[6];
    const float* b_out = (const float*)d_in[7];
    float* out = (float*)d_out;

    dim3 grid(BATCH / BM);      // 64 workgroups, each owns 16 batch rows
    dim3 block(NTHREADS);       // 8 waves of 32
    hipLaunchKernelGGL(gru_scan_kernel, grid, block, 0, stream,
                       input, h0, w_ih, w_hh, b_ih, b_hh, w_out, b_out, out);
}
